// AnyLocModel_21844203668357
// MI455X (gfx1250) — matrix operand order repaired and checked by reading, not compiled, not run on hardware
//
#include <hip/hip_runtime.h>
#include <hip/hip_bf16.h>

typedef __bf16 bf16;
typedef __attribute__((ext_vector_type(16))) __bf16 v16bf;
typedef __attribute__((ext_vector_type(8)))  __bf16 v8bf;
typedef __attribute__((ext_vector_type(4)))  __bf16 v4bf;
typedef __attribute__((ext_vector_type(8)))  float  v8f;

#define EPSN 1e-12f

// ---- WMMA fragment helpers (bf16, 16x16x32) -------------------------------
// A (16x32, row-major source, stride ld):
//   lanes 0-15  : row M=lane,    elems 0..15 -> K 0..15
//   lanes 16-31 : row M=lane-16, elems 0..15 -> K 16..31
__device__ __forceinline__ v16bf frag_a(const bf16* base, int ld, int lane) {
  const int m = lane & 15, kh = lane >> 4;
  return *(const v16bf*)(base + (size_t)m * ld + kh * 16);
}

// B (32x16): source is W-row-major [n][k] with stride ld.
//   lanes 0-15: col N=lane, K 0..15 ; lanes 16-31: col N=lane-16, K 16..31
__device__ __forceinline__ v16bf frag_b(const bf16* base, int ld, int lane) {
  const int n = lane & 15, kh = lane >> 4;
  return *(const v16bf*)(base + (size_t)n * ld + kh * 16);
}

__device__ __forceinline__ v8f wmma_bf16(v16bf a, v16bf b, v8f c) {
  return __builtin_amdgcn_wmma_f32_16x16x32_bf16(false, a, false, b,
                                                 (short)0, c, false, false);
}

__device__ __forceinline__ v8f vzero8() {
  v8f z;
#pragma unroll
  for (int j = 0; j < 8; ++j) z[j] = 0.f;
  return z;
}

// ---------------------------------------------------------------------------
// k1: h = l2norm(x @ pca_w^T + pca_b)   (32768 x 1024) x (1024 x 256)
// Software-pipelined: regs <- global(ks+1) overlaps WMMA(ks); double LDS buffers.
__global__ __launch_bounds__(512) void k1_pca(const float* __restrict__ x,
                                              const float* __restrict__ pca_w,
                                              const float* __restrict__ pca_b,
                                              bf16* __restrict__ h) {
  __shared__ bf16 la[2][64 * 32];
  __shared__ bf16 lb[2][256 * 32];
  __shared__ float rowsq[64];
  const int tid = threadIdx.x, lane = tid & 31, wave = tid >> 5;
  const int mt = wave >> 2, ng = wave & 3;
  const int tok0 = blockIdx.x * 64;
  if (tid < 64) rowsq[tid] = 0.f;

  const int ar = tid >> 3, akk = (tid & 7) * 4;   // A staging coords
  const int br = tid >> 1, bkk = (tid & 1) * 16;  // B staging coords
  const float* xa = x + (size_t)(tok0 + ar) * 1024 + akk;
  const float* wb = pca_w + (size_t)br * 1024 + bkk;

  float4 ra, rb[4];
  auto load_stage = [&](int ks) {
    const int k0 = ks * 32;
    ra = *(const float4*)(xa + k0);
#pragma unroll
    for (int i = 0; i < 4; ++i) rb[i] = *(const float4*)(wb + k0 + i * 4);
  };
  auto store_stage = [&](int buf) {
    bf16* A = la[buf] + ar * 32 + akk;
    A[0] = (bf16)ra.x; A[1] = (bf16)ra.y; A[2] = (bf16)ra.z; A[3] = (bf16)ra.w;
    bf16* B = lb[buf] + br * 32 + bkk;
#pragma unroll
    for (int i = 0; i < 4; ++i) {
      B[i * 4 + 0] = (bf16)rb[i].x; B[i * 4 + 1] = (bf16)rb[i].y;
      B[i * 4 + 2] = (bf16)rb[i].z; B[i * 4 + 3] = (bf16)rb[i].w;
    }
  };

  v8f acc[4];
#pragma unroll
  for (int i = 0; i < 4; ++i) acc[i] = vzero8();

  load_stage(0);
  for (int ks = 0; ks < 32; ++ks) {
    const int buf = ks & 1;
    store_stage(buf);
    __syncthreads();
    if (ks + 1 < 32) load_stage(ks + 1);   // overlaps with WMMA below
    const v16bf fa = frag_a(la[buf] + mt * 16 * 32, 32, lane);
#pragma unroll
    for (int i = 0; i < 4; ++i) {
      const v16bf fb = frag_b(lb[buf] + (ng * 4 + i) * 16 * 32, 32, lane);
      acc[i] = wmma_bf16(fa, fb, acc[i]);
    }
  }

  const int rh = lane >> 4;
#pragma unroll
  for (int i = 0; i < 4; ++i) {
    const float b = pca_b[(ng * 4 + i) * 16 + (lane & 15)];
#pragma unroll
    for (int j = 0; j < 8; ++j) acc[i][j] += b;
  }
  __syncthreads();  // all WMMA consumers done before rowsq accumulation begins
#pragma unroll
  for (int j = 0; j < 8; ++j) {
    float s = 0.f;
#pragma unroll
    for (int i = 0; i < 4; ++i) s += acc[i][j] * acc[i][j];
#pragma unroll
    for (int m = 8; m >= 1; m >>= 1) s += __shfl_xor(s, m, 32);
    if ((lane & 15) == 0) atomicAdd(&rowsq[mt * 16 + rh * 8 + j], s);
  }
  __syncthreads();
#pragma unroll
  for (int j = 0; j < 8; ++j) {
    const int mloc = mt * 16 + rh * 8 + j;
    const float inv = 1.f / fmaxf(sqrtf(rowsq[mloc]), EPSN);
#pragma unroll
    for (int i = 0; i < 4; ++i) {
      const int ncol = (ng * 4 + i) * 16 + (lane & 15);
      h[(size_t)(tok0 + mloc) * 256 + ncol] = (bf16)(acc[i][j] * inv);
    }
  }
}

// ---------------------------------------------------------------------------
// k2: a = softmax(h @ conv_w^T + conv_b), asum += sum_t a
__global__ __launch_bounds__(512) void k2_assign(const bf16* __restrict__ h,
                                                 const float* __restrict__ conv_w,
                                                 const float* __restrict__ conv_b,
                                                 bf16* __restrict__ a,
                                                 float* __restrict__ asum) {
  __shared__ bf16 cw[64 * 256];
  __shared__ float lg[64 * 64];
  const int tid = threadIdx.x, lane = tid & 31, wave = tid >> 5;
  const int mt = wave >> 2, nt = wave & 3;
  const int tok0 = blockIdx.x * 64;

  for (int i = tid; i < 64 * 256 / 4; i += 512) {  // stage conv_w f32->bf16
    const float4 v = *(const float4*)(conv_w + (size_t)i * 4);
    cw[i * 4 + 0] = (bf16)v.x; cw[i * 4 + 1] = (bf16)v.y;
    cw[i * 4 + 2] = (bf16)v.z; cw[i * 4 + 3] = (bf16)v.w;
  }
  __syncthreads();

  v8f acc = vzero8();
  const bf16* abase = h + (size_t)(tok0 + mt * 16) * 256;
#pragma unroll
  for (int ks = 0; ks < 8; ++ks) {
    const int k0 = ks * 32;
    if (ks + 1 < 8)  // pull next fragment line while this WMMA runs
      __builtin_prefetch(abase + (size_t)(lane & 15) * 256 + k0 + 32, 0, 0);
    const v16bf fa = frag_a(abase + k0, 256, lane);
    const v16bf fb = frag_b(cw + nt * 16 * 256 + k0, 256, lane);
    acc = wmma_bf16(fa, fb, acc);
  }
  const int rh = lane >> 4;
  const int ncol = nt * 16 + (lane & 15);
  const float b = conv_b[ncol];
#pragma unroll
  for (int j = 0; j < 8; ++j) lg[(mt * 16 + rh * 8 + j) * 64 + ncol] = acc[j] + b;
  __syncthreads();

  if (tid < 64) {  // softmax over 64 clusters, one row per thread
    float mx = -3.4e38f;
    for (int c = 0; c < 64; ++c) mx = fmaxf(mx, lg[tid * 64 + c]);
    float s = 0.f;
    for (int c = 0; c < 64; ++c) {
      const float e = __expf(lg[tid * 64 + c] - mx);
      lg[tid * 64 + c] = e; s += e;
    }
    const float inv = 1.f / s;
    for (int c = 0; c < 64; ++c) lg[tid * 64 + c] *= inv;
  }
  __syncthreads();
#pragma unroll
  for (int i = 0; i < 8; ++i) {
    const int idx = tid * 8 + i;
    a[(size_t)(tok0 + (idx >> 6)) * 64 + (idx & 63)] = (bf16)lg[idx];
  }
  if (tid < 64) {
    float s = 0.f;
    for (int t = 0; t < 64; ++t) s += lg[t * 64 + tid];
    atomicAdd(&asum[(tok0 >> 11) * 64 + tid], s);
  }
}

// ---------------------------------------------------------------------------
// k3: agg[n] += a[n]^T @ h[n]  (64 x 2048) x (2048 x 256)
// grid (16 batches, 8 K-splits); software-pipelined transposed staging.
__global__ __launch_bounds__(512) void k3_agg(const bf16* __restrict__ a,
                                              const bf16* __restrict__ h,
                                              float* __restrict__ agg) {
  __shared__ bf16 at[2][64 * 32];    // a transposed: [cluster][t]
  __shared__ bf16 ht[2][256 * 32];   // h transposed: [p][t]
  const int tid = threadIdx.x, lane = tid & 31, wave = tid >> 5;
  const int mt = wave >> 2, ng = wave & 3;
  const int n = blockIdx.x, kb = blockIdx.y;

  const int t = tid >> 4, c0 = (tid & 15) * 4, p0 = (tid & 15) * 16;
  const bf16* asrc = a + ((size_t)(n * 2048 + kb * 256 + t)) * 64 + c0;
  const bf16* hsrc = h + ((size_t)(n * 2048 + kb * 256 + t)) * 256 + p0;

  v4bf rA;
  v16bf rH;
  auto load_stage = [&](int ks) {
    rA = *(const v4bf*)(asrc + (size_t)ks * 32 * 64);
    rH = *(const v16bf*)(hsrc + (size_t)ks * 32 * 256);
  };
  auto store_stage = [&](int buf) {
#pragma unroll
    for (int i = 0; i < 4; ++i) at[buf][(c0 + i) * 32 + t] = rA[i];
#pragma unroll
    for (int i = 0; i < 16; ++i) ht[buf][(p0 + i) * 32 + t] = rH[i];
  };

  v8f acc[4];
#pragma unroll
  for (int i = 0; i < 4; ++i) acc[i] = vzero8();

  load_stage(0);
  for (int ks = 0; ks < 8; ++ks) {
    const int buf = ks & 1;
    store_stage(buf);
    __syncthreads();
    if (ks + 1 < 8) load_stage(ks + 1);
    const v16bf fa = frag_a(at[buf] + mt * 16 * 32, 32, lane);
#pragma unroll
    for (int i = 0; i < 4; ++i) {
      const v16bf fb = frag_b(ht[buf] + (ng * 4 + i) * 16 * 32, 32, lane);
      acc[i] = wmma_bf16(fa, fb, acc[i]);
    }
  }
  const int rh = lane >> 4;
#pragma unroll
  for (int i = 0; i < 4; ++i) {
    const int p = (ng * 4 + i) * 16 + (lane & 15);
#pragma unroll
    for (int j = 0; j < 8; ++j) {
      const int kc = mt * 16 + rh * 8 + j;
      atomicAdd(&agg[((size_t)n * 64 + kc) * 256 + p], acc[i][j]);
    }
  }
}

// ---------------------------------------------------------------------------
// k4: vlad = l2norm_global(l2norm_row(agg - asum*centroids)) -> bf16
__global__ __launch_bounds__(256) void k4_vlad(const float* __restrict__ agg,
                                               const float* __restrict__ asum,
                                               const float* __restrict__ cent,
                                               bf16* __restrict__ vlad) {
  __shared__ float rn[64];
  __shared__ float gn_s;
  const int tid = threadIdx.x, lane = tid & 31;
  const int n = blockIdx.x;

  for (int rr = 0; rr < 8; ++rr) {
    const int k = (tid >> 5) * 8 + rr;
    const float as = asum[n * 64 + k];
    float s = 0.f;
#pragma unroll
    for (int i = 0; i < 8; ++i) {
      const int p = lane * 8 + i;
      const float v = agg[((size_t)n * 64 + k) * 256 + p] - as * cent[k * 256 + p];
      s += v * v;
    }
#pragma unroll
    for (int m = 16; m >= 1; m >>= 1) s += __shfl_xor(s, m, 32);
    if (lane == 0) rn[k] = s;
  }
  __syncthreads();
  if (tid == 0) {
    float g = 0.f;
    for (int k = 0; k < 64; ++k) {
      const float nm = fmaxf(sqrtf(rn[k]), EPSN);
      g += rn[k] / (nm * nm);
    }
    gn_s = fmaxf(sqrtf(g), EPSN);
  }
  __syncthreads();
  const float ginv = 1.f / gn_s;
  for (int rr = 0; rr < 8; ++rr) {
    const int k = (tid >> 5) * 8 + rr;
    const float as = asum[n * 64 + k];
    const float sc = ginv / fmaxf(sqrtf(rn[k]), EPSN);
#pragma unroll
    for (int i = 0; i < 8; ++i) {
      const int p = lane * 8 + i;
      const float v = agg[((size_t)n * 64 + k) * 256 + p] - as * cent[k * 256 + p];
      vlad[(size_t)n * 16384 + k * 256 + p] = (bf16)(v * sc);
    }
  }
}

// ---------------------------------------------------------------------------
// k5: oacc += vlad @ comp_w^T   (16 x 16384) x (16384 x 2048), K split x32
// grid (16 col-groups of 128, 32 k-splits); software-pipelined staging.
__global__ __launch_bounds__(256) void k5_comp(const bf16* __restrict__ vlad,
                                               const float* __restrict__ comp_w,
                                               float* __restrict__ oacc) {
  __shared__ bf16 lb[2][128 * 32];
  const int tid = threadIdx.x, lane = tid & 31, wave = tid >> 5;
  const int g = blockIdx.x, kb = blockIdx.y;

  const int r = tid >> 1, kk = (tid & 1) * 16;
  const float* wsrc = comp_w + ((size_t)(g * 128 + r)) * 16384 + kb * 512 + kk;

  float4 rb[4];
  auto load_stage = [&](int ks) {
#pragma unroll
    for (int i = 0; i < 4; ++i) rb[i] = *(const float4*)(wsrc + ks * 32 + i * 4);
  };
  auto store_stage = [&](int buf) {
    bf16* B = lb[buf] + r * 32 + kk;
#pragma unroll
    for (int i = 0; i < 4; ++i) {
      B[i * 4 + 0] = (bf16)rb[i].x; B[i * 4 + 1] = (bf16)rb[i].y;
      B[i * 4 + 2] = (bf16)rb[i].z; B[i * 4 + 3] = (bf16)rb[i].w;
    }
  };

  v8f acc = vzero8();
  load_stage(0);
  for (int ks = 0; ks < 16; ++ks) {
    const int buf = ks & 1;
    store_stage(buf);
    __syncthreads();
    if (ks + 1 < 16) load_stage(ks + 1);
    const int k0 = kb * 512 + ks * 32;
    const v16bf fa = frag_a(vlad + k0, 16384, lane);
    const v16bf fb = frag_b(lb[buf] + wave * 16 * 32, 32, lane);
    acc = wmma_bf16(fa, fb, acc);
  }
  const int rh = lane >> 4;
  const int col = g * 128 + wave * 16 + (lane & 15);
#pragma unroll
  for (int j = 0; j < 8; ++j)
    atomicAdd(&oacc[(rh * 8 + j) * 2048 + col], acc[j]);
}

// ---------------------------------------------------------------------------
// k6: out = l2norm(oacc + comp_b) per row
__global__ __launch_bounds__(256) void k6_final(const float* __restrict__ oacc,
                                                const float* __restrict__ comp_b,
                                                float* __restrict__ out) {
  __shared__ float wsum[8];
  __shared__ float nrm;
  const int tid = threadIdx.x, lane = tid & 31, wave = tid >> 5;
  const int n = blockIdx.x;
  float v[8];
  float s = 0.f;
#pragma unroll
  for (int i = 0; i < 8; ++i) {
    const int c = tid + i * 256;
    v[i] = oacc[n * 2048 + c] + comp_b[c];
    s += v[i] * v[i];
  }
#pragma unroll
  for (int m = 16; m >= 1; m >>= 1) s += __shfl_xor(s, m, 32);
  if (lane == 0) wsum[wave] = s;
  __syncthreads();
  if (tid == 0) {
    float t = 0.f;
    for (int i = 0; i < 8; ++i) t += wsum[i];
    nrm = fmaxf(sqrtf(t), EPSN);
  }
  __syncthreads();
  const float inv = 1.f / nrm;
#pragma unroll
  for (int i = 0; i < 8; ++i) out[n * 2048 + tid + i * 256] = v[i] * inv;
}

// ---------------------------------------------------------------------------
extern "C" void kernel_launch(void* const* d_in, const int* in_sizes, int n_in,
                              void* d_out, int out_size, void* d_ws, size_t ws_size,
                              hipStream_t stream) {
  const float* x      = (const float*)d_in[0];
  const float* pca_w  = (const float*)d_in[1];
  const float* pca_b  = (const float*)d_in[2];
  const float* conv_w = (const float*)d_in[3];
  const float* conv_b = (const float*)d_in[4];
  const float* cent   = (const float*)d_in[5];
  const float* comp_w = (const float*)d_in[6];
  const float* comp_b = (const float*)d_in[7];
  float* out = (float*)d_out;

  char* ws = (char*)d_ws;
  size_t off = 0;
  auto alloc = [&](size_t bytes) -> void* {
    void* p = ws + off;
    off += (bytes + 255) & ~(size_t)255;
    return p;
  };
  bf16*  h    = (bf16*) alloc(32768ull * 256 * 2);   // 16 MB
  bf16*  a    = (bf16*) alloc(32768ull * 64 * 2);    // 4 MB
  float* asum = (float*)alloc(16ull * 64 * 4);
  float* agg  = (float*)alloc(16ull * 64 * 256 * 4); // 1 MB
  bf16*  vlad = (bf16*) alloc(16ull * 16384 * 2);    // 512 KB
  float* oacc = (float*)alloc(16ull * 2048 * 4);     // 128 KB

  hipMemsetAsync(asum, 0, 16ull * 64 * 4, stream);
  hipMemsetAsync(agg,  0, 16ull * 64 * 256 * 4, stream);
  hipMemsetAsync(oacc, 0, 16ull * 2048 * 4, stream);

  k1_pca   <<<512, 512, 0, stream>>>(x, pca_w, pca_b, h);
  k2_assign<<<512, 512, 0, stream>>>(h, conv_w, conv_b, a, asum);
  k3_agg   <<<dim3(16, 8), 512, 0, stream>>>(a, h, agg);
  k4_vlad  <<<16,  256, 0, stream>>>(agg, asum, cent, vlad);
  k5_comp  <<<dim3(16, 32), 256, 0, stream>>>(vlad, comp_w, oacc);
  k6_final <<<16,  256, 0, stream>>>(oacc, comp_b, out);
}
